// GraphModelSubstrate_30227979829530
// MI455X (gfx1250) — matrix operand order repaired and checked by reading, not compiled, not run on hardware
//
#include <hip/hip_runtime.h>
#include <math.h>

// ---------------- model constants (from reference) ----------------
constexpr int kG  = 512;
constexpr int kP  = 128;
constexpr int kC  = 256;
constexpr int kE  = 262144;
constexpr int kDS = 16;
constexpr int kDTR= 16;
constexpr int kL  = 2;
constexpr int kH  = 9;
constexpr int kN  = kG * kP;          // 65536
constexpr int k2C = 2 * kC;           // 512
constexpr float kEPS = 1e-5f;

typedef __attribute__((ext_vector_type(16))) _Float16 v16h;
typedef __attribute__((ext_vector_type(8)))  _Float16 h8;
typedef __attribute__((ext_vector_type(4)))  _Float16 h4;
typedef __attribute__((ext_vector_type(8)))  float    v8f;

// =================================================================
//  WMMA GEMM:  out[M,Nout] = act( (A (+Aadd)) @ W16^T + bias (+res) )
//  A: [M,K] f32 row-major; W16: [Nout,K] f16 row-major (pre-converted).
//  Requirements: K % 32 == 0, M % 128 == 0 (all call sites comply).
//  Block: 256 threads (8 waves). Block tile 128x64, wave tile 32x32.
//  Double-buffered LDS tiles: one barrier per K-step. Fragment loads
//  issue before next-tile staging so s_wait_dscnt covers only frags.
// =================================================================
constexpr int kAStr = 128 * 40;   // halves per A buffer (row stride 40)
constexpr int kBStr = 64 * 40;    // halves per B buffer

__device__ inline v16h frag_a(const _Float16* As, int row, int g8) {
  // A 16x32 f16 layout: lane&15 = M; lanes 0-15: K 0-7,16-23; lanes 16-31: K 8-15,24-31
  int base = row * 40 + g8;            // byte offset row*80 + g8*2 (16B aligned)
  union { v16h v; h8 p[2]; } u;
  u.p[0] = *reinterpret_cast<const h8*>(As + base);
  u.p[1] = *reinterpret_cast<const h8*>(As + base + 16);
  return u.v;
}
__device__ inline v16h frag_b(const _Float16* Bs, int col, int g16) {
  // B staged row-per-output-col (Bs[col][k]); lanes 0-15: K 0-7,16-23; lanes 16-31: K 8-15,24-31
  int base = col * 40 + g16;
  union { v16h v; h8 p[2]; } u;
  u.p[0] = *reinterpret_cast<const h8*>(Bs + base);
  u.p[1] = *reinterpret_cast<const h8*>(Bs + base + 16);
  return u.v;
}

template<bool HASADD, bool HASBIAS, bool HASRES, bool RELU>
__global__ __launch_bounds__(256)
void k_wmma_gemm(const float* __restrict__ A, const float* __restrict__ Aadd,
                 const _Float16* __restrict__ W, const float* __restrict__ bias,
                 const float* __restrict__ res, float* __restrict__ out,
                 int M, int K, int Nout) {
  __shared__ __align__(16) _Float16 As[2 * kAStr];
  __shared__ __align__(16) _Float16 Bs[2 * kBStr];

  const int tid  = threadIdx.x;
  const int lane = tid & 31;
  const int wave = tid >> 5;
  const int wr   = (wave >> 1) * 32;   // 0..96 (4 wave-rows)
  const int wc   = (wave & 1) * 32;    // 0..32 (2 wave-cols)
  const int l15  = lane & 15;
  const int g    = lane >> 4;
  const int g8   = g * 8;
  const int g16  = g * 16;
  const int mBlk = blockIdx.y * 128;
  const int nBlk = blockIdx.x * 64;

  v8f acc00 = {}, acc01 = {}, acc10 = {}, acc11 = {};
  const int nk = K >> 5;

  // ---- A stager: batch 4 float4 loads, then convert+store (h4) ----
  auto stageA = [&](int buf, int k0) {
    float4 ra[4];
#pragma unroll
    for (int i = 0; i < 4; ++i) {
      int idx4 = tid + i * 256;                 // 1024 float4 = 128x32
      int r = idx4 >> 3, kq = (idx4 & 7) * 4;
      ra[i] = *reinterpret_cast<const float4*>(&A[(size_t)(mBlk + r) * K + k0 + kq]);
    }
    if (HASADD) {
      float4 rb[4];
#pragma unroll
      for (int i = 0; i < 4; ++i) {
        int idx4 = tid + i * 256;
        int r = idx4 >> 3, kq = (idx4 & 7) * 4;
        rb[i] = *reinterpret_cast<const float4*>(&Aadd[(size_t)(mBlk + r) * K + k0 + kq]);
      }
#pragma unroll
      for (int i = 0; i < 4; ++i) {
        ra[i].x += rb[i].x; ra[i].y += rb[i].y; ra[i].z += rb[i].z; ra[i].w += rb[i].w;
      }
    }
#pragma unroll
    for (int i = 0; i < 4; ++i) {
      int idx4 = tid + i * 256;
      int r = idx4 >> 3, kq = (idx4 & 7) * 4;
      h4 hh = { (_Float16)ra[i].x, (_Float16)ra[i].y, (_Float16)ra[i].z, (_Float16)ra[i].w };
      *reinterpret_cast<h4*>(&As[buf * kAStr + r * 40 + kq]) = hh;
    }
  };
  // ---- B stager: weights already f16 -> one h8 load + b128 store ----
  auto stageB = [&](int buf, int k0) {
    int n = tid >> 2;                           // 64 rows x 4 chunks of 8 halves
    int kq = (tid & 3) * 8;
    int col = nBlk + n;
    h8 v = {};
    if (col < Nout) v = *reinterpret_cast<const h8*>(&W[(size_t)col * K + k0 + kq]);
    *reinterpret_cast<h8*>(&Bs[buf * kBStr + n * 40 + kq]) = v;
  };

  stageA(0, 0);
  stageB(0, 0);
  __syncthreads();

  for (int kt = 0; kt < nk; ++kt) {
    const int cur = kt & 1;
    const _Float16* Ab = As + cur * kAStr;
    const _Float16* Bb = Bs + cur * kBStr;

    // fragment ds_loads first (staging stores can drain under the WMMAs)
    v16h a0 = frag_a(Ab, wr + l15,      g8);
    v16h a1 = frag_a(Ab, wr + 16 + l15, g8);
    v16h b0 = frag_b(Bb, wc + l15,      g8);
    v16h b1 = frag_b(Bb, wc + 16 + l15, g8);

    if (kt + 1 < nk) {                 // stage next tile into other buffer
      stageA(cur ^ 1, (kt + 1) * 32);
      stageB(cur ^ 1, (kt + 1) * 32);
    }
    if (kt + 2 < nk) {                 // prefetch tile after next (global_prefetch_b8)
      __builtin_prefetch(&A[(size_t)(mBlk + (tid >> 1)) * K + (kt + 2) * 32], 0, 3);
      int pn = nBlk + (tid >> 2);
      if (pn < Nout) __builtin_prefetch(&W[(size_t)pn * K + (kt + 2) * 32], 0, 3);
    }

    acc00 = __builtin_amdgcn_wmma_f32_16x16x32_f16(false, a0, false, b0, (short)0, acc00, false, false);
    acc01 = __builtin_amdgcn_wmma_f32_16x16x32_f16(false, a0, false, b1, (short)0, acc01, false, false);
    acc10 = __builtin_amdgcn_wmma_f32_16x16x32_f16(false, a1, false, b0, (short)0, acc10, false, false);
    acc11 = __builtin_amdgcn_wmma_f32_16x16x32_f16(false, a1, false, b1, (short)0, acc11, false, false);
    __syncthreads();
  }

  // ---- epilogue: C/D 16x16 layout -> row = e + 8*(lane>>4), col = lane&15
  const v8f* accs[4] = { &acc00, &acc01, &acc10, &acc11 };
#pragma unroll
  for (int t = 0; t < 4; ++t) {
    int ti = t >> 1, tj = t & 1;
    const v8f a = *accs[t];
#pragma unroll
    for (int e = 0; e < 8; ++e) {
      int row = mBlk + wr + ti * 16 + g * 8 + e;
      int col = nBlk + wc + tj * 16 + l15;
      if (col < Nout) {
        float v = a[e];
        if (HASBIAS) v += bias[col];
        if (HASRES)  v += res[(size_t)row * Nout + col];
        if (RELU)    v = fmaxf(v, 0.f);
        out[(size_t)row * Nout + col] = v;
      }
    }
  }
}

// =================================================================
//  Helper kernels
// =================================================================
__global__ void k_zero(float* p, int n) {
  int i = blockIdx.x * 256 + threadIdx.x;
  if (i < n) p[i] = 0.f;
}

__global__ void k_add(const float* a, const float* b, float* o, int n) {
  int i = blockIdx.x * 256 + threadIdx.x;
  if (i < n) o[i] = a[i] + b[i];
}

// f32 -> f16 weight pre-conversion (vectorized), n4 = n/4
__global__ void k_cvt16(const float* __restrict__ src, _Float16* __restrict__ dst, int n4) {
  int i = blockIdx.x * 256 + threadIdx.x;
  if (i >= n4) return;
  float4 v = reinterpret_cast<const float4*>(src)[i];
  h4 h = { (_Float16)v.x, (_Float16)v.y, (_Float16)v.z, (_Float16)v.w };
  reinterpret_cast<h4*>(dst)[i] = h;
}

// small generic GEMM (odd K): out[m, coloff+o] = A@W^T + b
__global__ void k_sgemm_small(const float* __restrict__ A, int lda,
                              const float* __restrict__ W, const float* __restrict__ bias,
                              float* __restrict__ out, int ldo, int coloff,
                              int M, int K, int Nout) {
  int idx = blockIdx.x * 256 + threadIdx.x;
  if (idx >= M * Nout) return;
  int m = idx / Nout, o = idx % Nout;
  float s = bias ? bias[o] : 0.f;
  for (int k = 0; k < K; ++k) s += A[m * lda + k] * W[o * K + k];
  out[m * ldo + coloff + o] = s;
}

// pe batch statistics (3 columns over N rows), single block
__global__ void k_pe_stats(const float* __restrict__ pe, float* st) {
  __shared__ float sh[6 * 256];
  int tid = threadIdx.x;
  float s0 = 0, s1 = 0, s2 = 0, q0 = 0, q1 = 0, q2 = 0;
  for (int r = tid; r < kN; r += 256) {
    float a = pe[r * 3 + 0], b = pe[r * 3 + 1], c = pe[r * 3 + 2];
    s0 += a; s1 += b; s2 += c; q0 += a * a; q1 += b * b; q2 += c * c;
  }
  sh[0 * 256 + tid] = s0; sh[1 * 256 + tid] = s1; sh[2 * 256 + tid] = s2;
  sh[3 * 256 + tid] = q0; sh[4 * 256 + tid] = q1; sh[5 * 256 + tid] = q2;
  __syncthreads();
  if (tid < 6) {
    float acc = 0.f;
    for (int i = 0; i < 256; ++i) acc += sh[tid * 256 + i];
    st[tid] = acc;
  }
}

// pe: BN -> 3x3 linear -> xn[:,253:256]
__global__ void k_pe_embed(const float* __restrict__ pe, const float* __restrict__ st,
                           const float* __restrict__ bnw, const float* __restrict__ bnb,
                           const float* __restrict__ peW, const float* __restrict__ peb,
                           float* __restrict__ xn) {
  int n = blockIdx.x * 256 + threadIdx.x;
  if (n >= kN) return;
  float h[3];
#pragma unroll
  for (int k = 0; k < 3; ++k) {
    float m = st[k] * (1.f / kN);
    float v = st[3 + k] * (1.f / kN) - m * m;
    h[k] = (pe[n * 3 + k] - m) * rsqrtf(v + kEPS) * bnw[k] + bnb[k];
  }
#pragma unroll
  for (int j = 0; j < 3; ++j) {
    float s = peb[j];
#pragma unroll
    for (int k = 0; k < 3; ++k) s += h[k] * peW[j * 3 + k];
    xn[n * kC + (kC - 3) + j] = s;
  }
}

// GINE aggregation, edge embedding recomputed in-kernel (block = one edge)
__global__ void k_gine_agg(const float* __restrict__ xn, const float* __restrict__ eattr,
                           const float* __restrict__ eW, const float* __restrict__ eb,
                           const int* __restrict__ src, const int* __restrict__ dst,
                           float* __restrict__ agg) {
  __shared__ float ea[13];
  int e = blockIdx.x, c = threadIdx.x;
  if (c < 13) ea[c] = eattr[e * 13 + c];
  __syncthreads();
  float s = eb[c];
#pragma unroll
  for (int k = 0; k < 13; ++k) s += ea[k] * eW[c * 13 + k];
  float v = xn[src[e] * kC + c] + s;
  v = fmaxf(v, 0.f);
  atomicAdd(&agg[dst[e] * kC + c], v);
}

// causal depthwise conv(4) + SiLU; u = xz[:, 0:256]
__global__ void k_conv_silu(const float* __restrict__ xz, const float* __restrict__ cw,
                            const float* __restrict__ cb, float* __restrict__ u) {
  int idx = blockIdx.x * 256 + threadIdx.x;
  if (idx >= kN * kC) return;
  int n = idx >> 8, c = idx & 255;
  int g = n >> 7, p = n & 127;
  float s = cb[c];
#pragma unroll
  for (int j = 0; j < 4; ++j) {
    int pp = p - 3 + j;
    if (pp >= 0) s += xz[(g * kP + pp) * k2C + c] * cw[c * 4 + j];
  }
  u[idx] = s / (1.f + __expf(-s));
}

// delta = softplus(proj[:, :16] @ dt_W^T + dt_b)
__global__ void k_dt_softplus(const float* __restrict__ proj, const float* __restrict__ dtW,
                              const float* __restrict__ dtb, float* __restrict__ delta) {
  int idx = blockIdx.x * 256 + threadIdx.x;
  if (idx >= kN * kC) return;
  int n = idx >> 8, c = idx & 255;
  float s = dtb[c];
#pragma unroll
  for (int k = 0; k < kDTR; ++k) s += proj[n * 48 + k] * dtW[c * kDTR + k];
  delta[idx] = (s > 20.f) ? s : log1pf(__expf(s));
}

// selective scan: block = graph, thread = channel, state in registers
__global__ __launch_bounds__(256)
void k_scan(const float* __restrict__ proj, const float* __restrict__ delta,
            const float* __restrict__ u, const float* __restrict__ Alog,
            const float* __restrict__ Dp, float* __restrict__ y) {
  __shared__ float Bsh[kDS], Csh[kDS];
  int g = blockIdx.x, c = threadIdx.x;
  float Ac[kDS], s[kDS];
#pragma unroll
  for (int d = 0; d < kDS; ++d) { Ac[d] = -__expf(Alog[c * kDS + d]); s[d] = 0.f; }
  float Dc = Dp[c];
  for (int p = 0; p < kP; ++p) {
    __syncthreads();
    int n = g * kP + p;
    if (c < kDS)            Bsh[c]        = proj[n * 48 + 16 + c];
    else if (c < 2 * kDS)   Csh[c - kDS]  = proj[n * 48 + 32 + (c - kDS)];
    __syncthreads();
    float d  = delta[n * kC + c];
    float ut = u[n * kC + c];
    float du = d * ut;
    float yv = 0.f;
#pragma unroll
    for (int dd = 0; dd < kDS; ++dd) {
      s[dd] = __expf(d * Ac[dd]) * s[dd] + du * Bsh[dd];
      yv += s[dd] * Csh[dd];
    }
    y[n * kC + c] = yv + Dc * ut;
  }
}

// y *= silu(z), z = xz[:, 256:512]
__global__ void k_gate(const float* __restrict__ xz, float* __restrict__ y) {
  int idx = blockIdx.x * 256 + threadIdx.x;
  if (idx >= kN * kC) return;
  int n = idx >> 8, c = idx & 255;
  float z = xz[n * k2C + kC + c];
  y[idx] *= z / (1.f + __expf(-z));
}

// per-column sum/sumsq over Nrows (C=256 columns)
__global__ void k_bn_stats(const float* __restrict__ X, float* st, int Nrows) {
  int c = threadIdx.x;
  int r0 = blockIdx.x * 256;
  float s = 0.f, q = 0.f;
  for (int r = r0; r < r0 + 256 && r < Nrows; ++r) {
    float v = X[r * kC + c];
    s += v; q += v * v;
  }
  atomicAdd(&st[c], s);
  atomicAdd(&st[kC + c], q);
}

__global__ void k_bn_apply(const float* __restrict__ X, const float* __restrict__ st,
                           const float* __restrict__ w, const float* __restrict__ b,
                           float* __restrict__ out, float invN, int n, int relu) {
  int idx = blockIdx.x * 256 + threadIdx.x;
  if (idx >= n) return;
  int c = idx & 255;
  float m = st[c] * invN;
  float v = st[kC + c] * invN - m * m;
  float r = (X[idx] - m) * rsqrtf(v + kEPS) * w[c] + b[c];
  if (relu) r = fmaxf(r, 0.f);
  out[idx] = r;
}

// global add pool (uniform graphs)
__global__ void k_pool(const float* __restrict__ xn, float* __restrict__ pooled) {
  int g = blockIdx.x, c = threadIdx.x;
  float s = 0.f;
  for (int p = 0; p < kP; ++p) s += xn[(g * kP + p) * kC + c];
  pooled[g * kC + c] = s;
}

// per-row LayerNorm over C
__global__ void k_ln_head(const float* __restrict__ pooled, const float* __restrict__ w,
                          const float* __restrict__ b, float* __restrict__ out) {
  __shared__ float sh[256];
  int g = blockIdx.x, c = threadIdx.x;
  float x = pooled[g * kC + c];
  sh[c] = x; __syncthreads();
  for (int s = 128; s > 0; s >>= 1) { if (c < s) sh[c] += sh[c + s]; __syncthreads(); }
  float mean = sh[0] * (1.f / kC);
  __syncthreads();
  float d = x - mean;
  sh[c] = d * d; __syncthreads();
  for (int s = 128; s > 0; s >>= 1) { if (c < s) sh[c] += sh[c + s]; __syncthreads(); }
  float var = sh[0] * (1.f / kC);
  out[g * kC + c] = d * rsqrtf(var + kEPS) * w[c] + b[c];
}

// out[g] = dot(t[g,:], w2) + b2
__global__ void k_head_dot(const float* __restrict__ t, const float* __restrict__ w2,
                           const float* __restrict__ b2, float* __restrict__ out) {
  __shared__ float sh[256];
  int g = blockIdx.x, c = threadIdx.x;
  sh[c] = t[g * kC + c] * w2[c]; __syncthreads();
  for (int s = 128; s > 0; s >>= 1) { if (c < s) sh[c] += sh[c + s]; __syncthreads(); }
  if (c == 0) out[g] = sh[0] + b2[0];
}

// =================================================================
//  Orchestration
// =================================================================
static inline unsigned blks(int n) { return (unsigned)((n + 255) / 256); }

extern "C" void kernel_launch(void* const* d_in, const int* in_sizes, int n_in,
                              void* d_out, int out_size, void* d_ws, size_t ws_size,
                              hipStream_t stream) {
  (void)in_sizes; (void)n_in; (void)out_size; (void)ws_size;

  const float* x       = (const float*)d_in[0];
  const float* pe      = (const float*)d_in[1];
  const int*   eidx    = (const int*)  d_in[2];
  const float* eattr   = (const float*)d_in[4];
  const float* node_W  = (const float*)d_in[5];
  const float* node_b  = (const float*)d_in[6];
  const float* pe_bn_w = (const float*)d_in[7];
  const float* pe_bn_b = (const float*)d_in[8];
  const float* pe_W    = (const float*)d_in[9];
  const float* pe_b    = (const float*)d_in[10];
  const float* edge_W  = (const float*)d_in[11];
  const float* edge_b  = (const float*)d_in[12];
  const float* gW1     = (const float*)d_in[13];
  const float* gb1     = (const float*)d_in[14];
  const float* gW2     = (const float*)d_in[15];
  const float* gb2     = (const float*)d_in[16];
  const float* inW     = (const float*)d_in[17];
  const float* convW   = (const float*)d_in[18];
  const float* convB   = (const float*)d_in[19];
  const float* xprojW  = (const float*)d_in[20];
  const float* dtW     = (const float*)d_in[21];
  const float* dtB     = (const float*)d_in[22];
  const float* Alog    = (const float*)d_in[23];
  const float* Dp      = (const float*)d_in[24];
  const float* outW    = (const float*)d_in[25];
  const float* mW1     = (const float*)d_in[26];
  const float* mb1     = (const float*)d_in[27];
  const float* mW2     = (const float*)d_in[28];
  const float* mb2     = (const float*)d_in[29];
  const float* bn1w    = (const float*)d_in[30];
  const float* bn1b    = (const float*)d_in[31];
  const float* bn2w    = (const float*)d_in[32];
  const float* bn2b    = (const float*)d_in[33];
  const float* bn3w    = (const float*)d_in[34];
  const float* bn3b    = (const float*)d_in[35];
  const float* hlnw    = (const float*)d_in[36];
  const float* hlnb    = (const float*)d_in[37];
  const float* hW1     = (const float*)d_in[38];
  const float* hb1     = (const float*)d_in[39];
  const float* hbnw    = (const float*)d_in[40];
  const float* hbnb    = (const float*)d_in[41];
  const float* hW2     = (const float*)d_in[42];
  const float* hb2     = (const float*)d_in[43];
  float* out = (float*)d_out;

  // ---------------- workspace layout ----------------
  char* wsb = (char*)d_ws;
  const size_t NB = (size_t)kN * kC * sizeof(float);   // 64 MiB
  float* xn    = (float*)(wsb);                 // [N,C]
  float* agg   = (float*)(wsb + 1 * NB);        // [N,C]
  float* t0    = (float*)(wsb + 2 * NB);        // [N,C]
  float* hbuf  = (float*)(wsb + 3 * NB);        // [N,C]
  float* h1    = (float*)(wsb + 4 * NB);        // [N,C]
  float* xz    = (float*)(wsb + 5 * NB);        // [N,2C]
  float* uconv = (float*)(wsb + 7 * NB);        // [N,C]
  float* delta = (float*)(wsb + 8 * NB);        // [N,C]
  float* ybuf  = (float*)(wsb + 9 * NB);        // [N,C]
  float* proj  = (float*)(wsb + 10 * NB);       // [N,48]
  char*  tail  = wsb + 10 * NB + (size_t)kN * 48 * sizeof(float);
  float* stats   = (float*)tail;                       // [2C]
  float* pestats = (float*)(tail + 2 * kC * 4);        // [6]
  float* pooled  = (float*)(tail + 2 * kC * 4 + 64);   // [G,C]
  float* th1     = pooled + kG * kC;                   // [G,C]
  float* th2     = th1 + kG * kC;                      // [G,C]

  // f16 pre-converted weights (16B-aligned: all preceding offsets are)
  _Float16* w16 = (_Float16*)(th2 + kG * kC);
  const int nGW1 = kL * kC * kC;      // 131072
  const int nGW2 = kL * kC * kC;
  const int nInW = kL * k2C * kC;     // 262144
  const int nXpr = kL * 48 * kC;      // 24576
  const int nOutW= kL * kC * kC;
  const int nMW1 = kL * k2C * kC;
  const int nMW2 = kL * kC * k2C;
  const int nHW1 = kH * kC * kC;      // 589824
  _Float16* gW1_16 = w16;
  _Float16* gW2_16 = gW1_16 + nGW1;
  _Float16* inW_16 = gW2_16 + nGW2;
  _Float16* xpr_16 = inW_16 + nInW;
  _Float16* outW_16= xpr_16 + nXpr;
  _Float16* mW1_16 = outW_16 + nOutW;
  _Float16* mW2_16 = mW1_16 + nMW1;
  _Float16* hW1_16 = mW2_16 + nMW2;

  const dim3 blk256(256);
  const float invN = 1.f / kN;
  const dim3 gNC(kC / 64, kN / 128);    // [N,C] output GEMMs
  const dim3 gN2C(k2C / 64, kN / 128);  // [N,2C] output GEMMs
  const dim3 gNP(1, kN / 128);          // [N,48] x-proj
  const dim3 gHC(kC / 64, kG / 128);    // head GEMMs [G,C]

  // ---------- one-shot weight f32->f16 conversion ----------
  k_cvt16<<<blks(nGW1 / 4), blk256, 0, stream>>>(gW1,    gW1_16, nGW1 / 4);
  k_cvt16<<<blks(nGW2 / 4), blk256, 0, stream>>>(gW2,    gW2_16, nGW2 / 4);
  k_cvt16<<<blks(nInW / 4), blk256, 0, stream>>>(inW,    inW_16, nInW / 4);
  k_cvt16<<<blks(nXpr / 4), blk256, 0, stream>>>(xprojW, xpr_16, nXpr / 4);
  k_cvt16<<<blks(nOutW/ 4), blk256, 0, stream>>>(outW,   outW_16,nOutW/ 4);
  k_cvt16<<<blks(nMW1 / 4), blk256, 0, stream>>>(mW1,    mW1_16, nMW1 / 4);
  k_cvt16<<<blks(nMW2 / 4), blk256, 0, stream>>>(mW2,    mW2_16, nMW2 / 4);
  k_cvt16<<<blks(nHW1 / 4), blk256, 0, stream>>>(hW1,    hW1_16, nHW1 / 4);

  // ---------- embeddings ----------
  k_pe_stats<<<1, blk256, 0, stream>>>(pe, pestats);
  k_sgemm_small<<<blks(kN * (kC - 3)), blk256, 0, stream>>>(
      x, 173, node_W, node_b, xn, kC, 0, kN, 173, kC - 3);
  k_pe_embed<<<blks(kN), blk256, 0, stream>>>(pe, pestats, pe_bn_w, pe_bn_b, pe_W, pe_b, xn);

  // ---------- layers ----------
  for (int l = 0; l < kL; ++l) {
    const _Float16* gW1l = gW1_16 + (size_t)l * kC * kC;
    const _Float16* gW2l = gW2_16 + (size_t)l * kC * kC;
    const _Float16* inWl = inW_16 + (size_t)l * k2C * kC;
    const _Float16* xprl = xpr_16 + (size_t)l * 48 * kC;
    const _Float16* outWl= outW_16+ (size_t)l * kC * kC;
    const _Float16* mW1l = mW1_16 + (size_t)l * k2C * kC;
    const _Float16* mW2l = mW2_16 + (size_t)l * kC * k2C;

    // GINE: agg then nn(x + agg), residual fused into 2nd GEMM
    k_zero<<<blks(kN * kC), blk256, 0, stream>>>(agg, kN * kC);
    k_gine_agg<<<kE, blk256, 0, stream>>>(xn, eattr, edge_W, edge_b, eidx, eidx + kE, agg);
    k_wmma_gemm<true,  true,  false, true ><<<gNC, blk256, 0, stream>>>(
        xn, agg, gW1l, gb1 + l * kC, nullptr, t0, kN, kC, kC);
    k_wmma_gemm<false, true,  true,  false><<<gNC, blk256, 0, stream>>>(
        t0, nullptr, gW2l, gb2 + l * kC, xn, hbuf, kN, kC, kC);
    k_zero<<<blks(2 * kC), blk256, 0, stream>>>(stats, 2 * kC);
    k_bn_stats<<<blks(kN), blk256, 0, stream>>>(hbuf, stats, kN);
    k_bn_apply<<<blks(kN * kC), blk256, 0, stream>>>(
        hbuf, stats, bn1w + l * kC, bn1b + l * kC, h1, invN, kN * kC, 0);

    // Mamba
    k_wmma_gemm<false, false, false, false><<<gN2C, blk256, 0, stream>>>(
        xn, nullptr, inWl, nullptr, nullptr, xz, kN, kC, k2C);
    k_conv_silu<<<blks(kN * kC), blk256, 0, stream>>>(xz, convW + l * kC * 4, convB + l * kC, uconv);
    k_wmma_gemm<false, false, false, false><<<gNP, blk256, 0, stream>>>(
        uconv, nullptr, xprl, nullptr, nullptr, proj, kN, kC, 48);
    k_dt_softplus<<<blks(kN * kC), blk256, 0, stream>>>(
        proj, dtW + (size_t)l * kC * kDTR, dtB + l * kC, delta);
    k_scan<<<kG, blk256, 0, stream>>>(
        proj, delta, uconv, Alog + (size_t)l * kC * kDS, Dp + l * kC, ybuf);
    k_gate<<<blks(kN * kC), blk256, 0, stream>>>(xz, ybuf);
    k_wmma_gemm<false, false, true,  false><<<gNC, blk256, 0, stream>>>(
        ybuf, nullptr, outWl, nullptr, xn, hbuf, kN, kC, kC);
    k_zero<<<blks(2 * kC), blk256, 0, stream>>>(stats, 2 * kC);
    k_bn_stats<<<blks(kN), blk256, 0, stream>>>(hbuf, stats, kN);
    k_bn_apply<<<blks(kN * kC), blk256, 0, stream>>>(
        hbuf, stats, bn2w + l * kC, bn2b + l * kC, hbuf, invN, kN * kC, 0);

    // combine + MLP (residual fused) + BN3 -> xn
    k_add<<<blks(kN * kC), blk256, 0, stream>>>(h1, hbuf, t0, kN * kC);
    k_wmma_gemm<false, true,  false, true ><<<gN2C, blk256, 0, stream>>>(
        t0, nullptr, mW1l, mb1 + l * k2C, nullptr, xz, kN, kC, k2C);
    k_wmma_gemm<false, true,  true,  false><<<gNC, blk256, 0, stream>>>(
        xz, nullptr, mW2l, mb2 + l * kC, t0, hbuf, kN, k2C, kC);
    k_zero<<<blks(2 * kC), blk256, 0, stream>>>(stats, 2 * kC);
    k_bn_stats<<<blks(kN), blk256, 0, stream>>>(hbuf, stats, kN);
    k_bn_apply<<<blks(kN * kC), blk256, 0, stream>>>(
        hbuf, stats, bn3w + l * kC, bn3b + l * kC, xn, invN, kN * kC, 0);
  }

  // ---------- pool + heads ----------
  k_pool<<<kG, blk256, 0, stream>>>(xn, pooled);
  for (int h = 0; h < kH; ++h) {
    k_ln_head<<<kG, blk256, 0, stream>>>(pooled, hlnw + h * kC, hlnb + h * kC, th1);
    k_wmma_gemm<false, true,  false, false><<<gHC, blk256, 0, stream>>>(
        th1, nullptr, hW1_16 + (size_t)h * kC * kC, hb1 + h * kC, nullptr, th2, kG, kC, kC);
    k_zero<<<blks(2 * kC), blk256, 0, stream>>>(stats, 2 * kC);
    k_bn_stats<<<blks(kG), blk256, 0, stream>>>(th2, stats, kG);
    k_bn_apply<<<blks(kG * kC), blk256, 0, stream>>>(
        th2, stats, hbnw + h * kC, hbnb + h * kC, th2, 1.f / kG, kG * kC, 1);
    k_head_dot<<<kG, blk256, 0, stream>>>(th2, hW2 + h * kC, hb2 + h, out + h * kG);
  }
}
